// RobustTrajectoryKAN_56959856279930
// MI455X (gfx1250) — matrix operand order repaired and checked by reading, not compiled, not run on hardware
//
#include <hip/hip_runtime.h>

typedef _Float16 v16h  __attribute__((ext_vector_type(16)));
typedef _Float16 v8h   __attribute__((ext_vector_type(8)));
typedef float    v8f   __attribute__((ext_vector_type(8)));
typedef int      i32x4 __attribute__((vector_size(16)));

#define AS_GLOBAL __attribute__((address_space(1)))
#define AS_SHARED __attribute__((address_space(3)))

#define BATCH      131072
#define M_TILE     128
#define LDS_STRIDE 40   // halves per LDS row: 32 data + 8 pad (20 dwords -> bank spread)

#if defined(__has_builtin)
#if __has_builtin(__builtin_amdgcn_global_load_async_to_lds_b128)
#define HAVE_ASYNC_B128 1
#endif
#endif

// ---------------------------------------------------------------------------
// Augmented f16 weights: W[o, i*16+0] = bw[o,i]; W[o, i*16+1+c] = sw[o,i,c]*ss[o,i]
// (c=0..12); slots 14,15 zero (K padding). Rows o >= out_real zero (N padding).
// ---------------------------------------------------------------------------
__global__ __launch_bounds__(256) void kan_prep_waug(
    const float* __restrict__ bw, const float* __restrict__ sw,
    const float* __restrict__ ss, _Float16* __restrict__ waug,
    int out_real, int out_pad, int in_feats)
{
  int idx = blockIdx.x * blockDim.x + threadIdx.x;
  int total = out_pad * in_feats;
  if (idx >= total) return;
  int o = idx / in_feats;
  int i = idx - o * in_feats;
  _Float16* dst = waug + (size_t)idx * 16;
  if (o < out_real) {
    float scale = ss[(size_t)o * in_feats + i];
    dst[0] = (_Float16)bw[(size_t)o * in_feats + i];
#pragma unroll
    for (int c = 0; c < 13; ++c)
      dst[1 + c] = (_Float16)(sw[((size_t)o * in_feats + i) * 13 + c] * scale);
    dst[14] = (_Float16)0.0f;
    dst[15] = (_Float16)0.0f;
  } else {
#pragma unroll
    for (int c = 0; c < 16; ++c) dst[c] = (_Float16)0.0f;
  }
}

// ---------------------------------------------------------------------------
// Fused KAN layer. Per 32-wide K chunk (double-buffered LDS, one barrier):
//   * 256 threads emit silu + the 4 nonzero closed-form cubic B-spline bases
//     (uniform grid h=0.2) straight into the f16 A tile in LDS,
//   * the 8 KB B chunk is async-copied global->LDS once per workgroup
//     (kills the 8x per-wave redundant global B reads),
//   * all 8 waves run v_wmma_f32_16x16x32_f16 from LDS; all NT B fragments
//     are loaded before the WMMA chain so dscnt waits form a staircase.
// ---------------------------------------------------------------------------
template <int IN, int NT>
__global__ __launch_bounds__(256) void kan_layer(
    const float* __restrict__ x, const _Float16* __restrict__ waug,
    float* __restrict__ y, int out_real)
{
  constexpr int K  = IN * 16;   // augmented K dimension
  constexpr int KC = K / 32;    // number of 32-wide K chunks

  __shared__ _Float16 ldsA[2 * M_TILE * LDS_STRIDE];
  __shared__ _Float16 ldsB[2 * M_TILE * LDS_STRIDE];

  const int tid   = threadIdx.x;
  const int lane  = tid & 31;
  const int wave  = tid >> 5;        // 0..7
  const int hi    = lane >> 4;       // half-wave select
  const int r     = lane & 15;       // row (A) / col (B,D) within tile
  const int rows0 = blockIdx.x * M_TILE;
  const int m0    = wave * 16;       // wave's 16-row strip within the block

  v8f acc[NT] = {};

  const int prow = tid >> 1;         // producer: row 0..127
  const int psel = tid & 1;          // producer: which of the 2 features

  for (int kc = 0; kc < KC; ++kc) {
    const int buf = (kc & 1) * (M_TILE * LDS_STRIDE);

    // -------- stage B chunk (NT*16 rows x 32 halves) into LDS -------------
    {
      const int btotal = NT * 16 * 4;          // # of 16-byte segments
#pragma unroll
      for (int rep = 0; rep < 2; ++rep) {
        const int c = tid + rep * 256;
        if (c < btotal) {
          const int o   = c >> 2;
          const int seg = c & 3;
          const _Float16* gsrc = waug + (size_t)o * K + (size_t)kc * 32 + seg * 8;
          _Float16*       ldst = &ldsB[buf + o * LDS_STRIDE + seg * 8];
#if defined(HAVE_ASYNC_B128)
          __builtin_amdgcn_global_load_async_to_lds_b128(
              (AS_GLOBAL i32x4*)gsrc,
              (AS_SHARED i32x4*)ldst, 0, 0);
#else
          *(v8h*)ldst = *(const v8h*)gsrc;
#endif
        }
      }
    }

    // -------- produce augmented activation chunk -> LDS -------------------
    {
      const int   pf = kc * 2 + psel;                    // feature index
      const float xv = x[(size_t)(rows0 + prow) * IN + pf];
      const float sil = xv * __builtin_amdgcn_rcpf(1.0f + __expf(-xv));

      _Float16* dst = &ldsA[buf + prow * LDS_STRIDE + psel * 16];
      v8h z = {};
      v8h p0 = z;
      p0[0] = (_Float16)sil;
      *(v8h*)dst       = p0;
      *(v8h*)(dst + 8) = z;

      // closed-form cubic B-spline on uniform knots g(t) = -1 + (t-3)*0.2:
      // box j with local coord f; nonzero bases at slots j-3..j (clipped 0..12)
      const float u  = (xv + 1.0f) * 5.0f;
      const float fu = floorf(u);
      const int   j  = (int)fu + 3;
      if (j >= 0 && j <= 15) {
        const float f   = u - fu;
        const float omf = 1.0f - f;
        const float f2  = f * f;
        const float f3  = f2 * f;
        const float w0  = omf * omf * omf * (1.0f / 6.0f);
        const float w1  = (3.0f * f3 - 6.0f * f2 + 4.0f) * (1.0f / 6.0f);
        const float w2  = (-3.0f * f3 + 3.0f * f2 + 3.0f * f + 1.0f) * (1.0f / 6.0f);
        const float w3  = f3 * (1.0f / 6.0f);
        if (j >= 3)             dst[j - 2] = (_Float16)w0;  // slot 1+(j-3)
        if (j >= 2 && j <= 14)  dst[j - 1] = (_Float16)w1;  // slot 1+(j-2)
        if (j >= 1 && j <= 13)  dst[j]     = (_Float16)w2;  // slot 1+(j-1)
        if (j <= 12)            dst[j + 1] = (_Float16)w3;  // slot 1+j
      }
    }

#if defined(HAVE_ASYNC_B128)
#if __has_builtin(__builtin_amdgcn_s_wait_asynccnt)
    __builtin_amdgcn_s_wait_asynccnt(0);
#else
    asm volatile("s_wait_asynccnt 0" ::: "memory");
#endif
#endif
    __syncthreads();   // single barrier per chunk; LDS double-buffered

    // -------- consume: WMMA over the 32-wide K chunk ----------------------
    // A fragment (16x32 f16): element e -> K = e + 8*(e>=8) + 8*hi
    const _Float16* ap = &ldsA[buf + (m0 + r) * LDS_STRIDE];
    const v8h alo = *(const v8h*)(ap + hi * 8);
    const v8h ahi = *(const v8h*)(ap + 16 + hi * 8);
    v16h a;
#pragma unroll
    for (int jj = 0; jj < 8; ++jj) { a[jj] = alo[jj]; a[jj + 8] = ahi[jj]; }

    // B fragments (32x16 f16): element e -> K = e + 8*(e>=8) + 8*hi, col = lane&15.
    // Load ALL fragments first so the WMMA chain runs without full drains.
    const _Float16* bb = &ldsB[buf];
    v16h bf[NT];
#pragma unroll
    for (int nt = 0; nt < NT; ++nt) {
      const _Float16* bp = bb + (nt * 16 + r) * LDS_STRIDE + hi * 8;
      const v8h b0 = *(const v8h*)bp;
      const v8h b1 = *(const v8h*)(bp + 16);
#pragma unroll
      for (int jj = 0; jj < 8; ++jj) { bf[nt][jj] = b0[jj]; bf[nt][jj + 8] = b1[jj]; }
    }
#pragma unroll
    for (int nt = 0; nt < NT; ++nt) {
      acc[nt] = __builtin_amdgcn_wmma_f32_16x16x32_f16(
          false, a, false, bf[nt], (short)0, acc[nt], false, false);
    }
  }

  // -------- store D: col = lane&15, row = vgpr + 8*hi ---------------------
#pragma unroll
  for (int nt = 0; nt < NT; ++nt) {
    const int col = nt * 16 + r;
    if (col < out_real) {
#pragma unroll
      for (int v = 0; v < 8; ++v) {
        const int m = hi * 8 + v;
        y[(size_t)(rows0 + m0 + m) * out_real + col] = acc[nt][v];
      }
    }
  }
}

// ---------------------------------------------------------------------------
extern "C" void kernel_launch(void* const* d_in, const int* in_sizes, int n_in,
                              void* d_out, int out_size, void* d_ws, size_t ws_size,
                              hipStream_t stream) {
  (void)in_sizes; (void)n_in; (void)out_size; (void)ws_size;

  const float* x   = (const float*)d_in[0];
  const float* bw0 = (const float*)d_in[1];
  const float* sw0 = (const float*)d_in[2];
  const float* ss0 = (const float*)d_in[3];
  const float* bw1 = (const float*)d_in[4];
  const float* sw1 = (const float*)d_in[5];
  const float* ss1 = (const float*)d_in[6];
  const float* bw2 = (const float*)d_in[7];
  const float* sw2 = (const float*)d_in[8];
  const float* ss2 = (const float*)d_in[9];
  const float* bw3 = (const float*)d_in[10];
  const float* sw3 = (const float*)d_in[11];
  const float* ss3 = (const float*)d_in[12];

  // workspace: two f32 activation ping-pong buffers + f16 augmented weights
  char*  ws   = (char*)d_ws;
  const size_t actBytes = (size_t)BATCH * 128 * sizeof(float);  // 64 MB
  float* buf0 = (float*)ws;
  float* buf1 = (float*)(ws + actBytes);
  char*  wb   = ws + 2 * actBytes;
  _Float16* w0 = (_Float16*)wb;                                      // 128 x 128
  _Float16* w1 = (_Float16*)(wb + (size_t)128 * 8   * 16 * 2);       // 128 x 2048
  _Float16* w2 = (_Float16*)(wb + (size_t)128 * 8   * 16 * 2
                                 + (size_t)128 * 128 * 16 * 2);      // 128 x 2048
  _Float16* w3 = (_Float16*)(wb + (size_t)128 * 8   * 16 * 2
                                 + (size_t)128 * 128 * 16 * 2 * 2);  // 16  x 2048

  kan_prep_waug<<<(128 * 8   + 255) / 256, 256, 0, stream>>>(bw0, sw0, ss0, w0, 128, 128, 8);
  kan_prep_waug<<<(128 * 128 + 255) / 256, 256, 0, stream>>>(bw1, sw1, ss1, w1, 128, 128, 128);
  kan_prep_waug<<<(128 * 128 + 255) / 256, 256, 0, stream>>>(bw2, sw2, ss2, w2, 128, 128, 128);
  kan_prep_waug<<<(16  * 128 + 255) / 256, 256, 0, stream>>>(bw3, sw3, ss3, w3, 2,   16,  128);

  const int nblk = BATCH / M_TILE;  // 1024 workgroups, 8 waves each
  kan_layer<8,   8><<<nblk, 256, 0, stream>>>(x,    w0, buf0, 128);
  kan_layer<128, 8><<<nblk, 256, 0, stream>>>(buf0, w1, buf1, 128);
  kan_layer<128, 8><<<nblk, 256, 0, stream>>>(buf1, w2, buf0, 128);
  kan_layer<128, 1><<<nblk, 256, 0, stream>>>(buf0, w3, (float*)d_out, 2);
}
